// _HGPT_4131758539346
// MI455X (gfx1250) — matrix operand order repaired and checked by reading, not compiled, not run on hardware
//
#include <hip/hip_runtime.h>
#include <hip/hip_bf16.h>
#include <math.h>

// Model constants (from reference)
#define Bc    2
#define Tc    1024
#define Vc    50257
#define Lc    8
#define Dc    1024
#define Hc    16
#define HKVc  4
#define HDc   64
#define KVDc  256
#define MLPc  4096
#define BDc   256
#define VEDc  128
#define NTOK  (Bc*Tc)          // 2048 rows

typedef __bf16 bf16;
typedef __attribute__((ext_vector_type(16))) __bf16 bf16x16;
typedef __attribute__((ext_vector_type(8)))  __bf16 bf16x8;
typedef __attribute__((ext_vector_type(4)))  __bf16 bf16x4;
typedef __attribute__((ext_vector_type(8)))  float  f32x8;
typedef __attribute__((address_space(3))) char lds_char;

__device__ __forceinline__ f32x8 wmma_bf16(bf16x16 a, bf16x16 b, f32x8 c) {
  return __builtin_amdgcn_wmma_f32_16x16x32_bf16(false, a, false, b, (short)0, c,
                                                 false, false);
}

__device__ __forceinline__ bf16 f2bf(float f) { return (bf16)f; }

// ---- gfx1250 async global->LDS copy (16 bytes/lane, tracked by ASYNCcnt) ----
__device__ __forceinline__ void async_copy16(void* ldst, const void* gsrc) {
  unsigned l = (unsigned)(size_t)(lds_char*)ldst;          // LDS byte offset
  asm volatile("global_load_async_to_lds_b128 %0, %1, off"
               :: "v"(l), "v"((unsigned long long)(size_t)gsrc)
               : "memory");
}
__device__ __forceinline__ void wait_async0() {
  asm volatile("s_wait_asynccnt 0x0" ::: "memory");
}

// ---- WMMA fragment loaders (wave32, 16x16x32 bf16) --------------------------
// A (16x32, MxK): lane row m = lane&15 ; K chunks [kb..kb+7] and [kb+16..kb+23],
// kb = 8*(lane>>4).  Two 16B LDS loads.
__device__ __forceinline__ bf16x16 load_fragA(const bf16* base, int row0, int ld) {
  int lane = threadIdx.x & 31;
  const bf16* p = base + (size_t)(row0 + (lane & 15)) * ld + ((lane >> 4) * 8);
  bf16x8 lo = *(const bf16x8*)p;
  bf16x8 hi = *(const bf16x8*)(p + 16);
  return __builtin_shufflevector(lo, hi, 0,1,2,3,4,5,6,7,8,9,10,11,12,13,14,15);
}
// B (32x16, KxN): lane column n = lane&15 ; K chunks [kb..kb+7] and [kb+16..kb+23], kb = 8*(lane>>4).
__device__ __forceinline__ bf16x16 load_fragB(const bf16* base, int row0, int ld) {
  int lane = threadIdx.x & 31;
  const bf16* p = base + (size_t)(row0 + (lane & 15)) * ld + ((lane >> 4) * 8);
  bf16x8 lo = *(const bf16x8*)p;
  bf16x8 hi = *(const bf16x8*)(p + 16);
  return __builtin_shufflevector(lo, hi, 0,1,2,3,4,5,6,7,8,9,10,11,12,13,14,15);
}

// ---- f32 -> bf16 weight conversion (once per weight per launch) -------------
__global__ __launch_bounds__(256)
void k_cvt(const float* __restrict__ src, bf16* __restrict__ dst, long n4)
{
  long i = (long)blockIdx.x * 256 + threadIdx.x;   // in float4 units
  if (i >= n4) return;
  float4 v = ((const float4*)src)[i];
  bf16x4 o; o[0] = f2bf(v.x); o[1] = f2bf(v.y); o[2] = f2bf(v.z); o[3] = f2bf(v.w);
  *(bf16x4*)(dst + i * 4) = o;
}

// ---- Generic GEMM:  C[M,N] = epilogue( A[M,K] @ W[N,K]^T ), A/W bf16 --------
// 256 threads = 8 waves (4x2); block tile 128x128, K-step 32.  Both operand
// tiles are staged with async global->LDS copies, double-buffered; the copy
// for tile k+1 overlaps WMMA on tile k; one s_wait_asynccnt + barrier per tile.
#define GM 128
#define GN 128
#define GK 32
#define LDT 40   // padded bf16 stride (conflict-free b128 reads)

enum { EP_NONE = 0, EP_SCALAR = 1, EP_BASE_SCALAR = 2, EP_BASE_COL = 3,
       EP_ADD = 4, EP_SQRELU = 5 };

__global__ __launch_bounds__(256)
void k_gemm(const bf16* __restrict__ A, const bf16* __restrict__ W,
            void* __restrict__ Cv, int M, int N, int K, int ep, int obf,
            const float* __restrict__ base, const float* __restrict__ colscale,
            const float* __restrict__ scalar)
{
  __shared__ __align__(16) bf16 As[2][GM * LDT];
  __shared__ __align__(16) bf16 Bs[2][GN * LDT];
  const int m0  = blockIdx.y * GM;
  const int n0  = blockIdx.x * GN;
  const int tid = threadIdx.x;
  const int wave = tid >> 5;
  const int wm = (wave >> 1) * 32;   // 0,32,64,96
  const int wn = (wave & 1) * 64;    // 0,64

  f32x8 acc[2][4] = {};

  auto issue_tile = [&](int kk, int buf) {
#pragma unroll
    for (int i = 0; i < 2; ++i) {    // A tile 128x32: 512 16B chunks
      int vid = tid + i * 256;
      int row = vid >> 2;
      int col = (vid & 3) * 8;
      async_copy16(&As[buf][row * LDT + col],
                   A + (size_t)(m0 + row) * K + kk + col);
    }
#pragma unroll
    for (int i = 0; i < 2; ++i) {    // W tile 128x32: rows past N are clamped;
      int vid = tid + i * 256;       // garbage only feeds masked-out columns.
      int row = vid >> 2;
      int col = (vid & 3) * 8;
      int rsrc = n0 + row; if (rsrc >= N) rsrc = N - 1;
      async_copy16(&Bs[buf][row * LDT + col],
                   W + (size_t)rsrc * K + kk + col);
    }
  };

  issue_tile(0, 0);
  wait_async0();
  __syncthreads();

  int cur = 0;
  for (int kk = 0; kk < K; kk += GK) {
    const bool more = (kk + GK) < K;
    if (more) issue_tile(kk + GK, cur ^ 1);   // overlap copy with WMMA below

    // hoist ALL fragment loads so the ds_load clause issues ahead of the
    // wmma chain (staggered dscnt waits instead of lockstep load->wmma)
    bf16x16 a0 = load_fragA(As[cur], wm, LDT);
    bf16x16 a1 = load_fragA(As[cur], wm + 16, LDT);
    bf16x16 b0 = load_fragB(Bs[cur], wn, LDT);
    bf16x16 b1 = load_fragB(Bs[cur], wn + 16, LDT);
    bf16x16 b2 = load_fragB(Bs[cur], wn + 32, LDT);
    bf16x16 b3 = load_fragB(Bs[cur], wn + 48, LDT);
    acc[0][0] = wmma_bf16(a0, b0, acc[0][0]);
    acc[1][0] = wmma_bf16(a1, b0, acc[1][0]);
    acc[0][1] = wmma_bf16(a0, b1, acc[0][1]);
    acc[1][1] = wmma_bf16(a1, b1, acc[1][1]);
    acc[0][2] = wmma_bf16(a0, b2, acc[0][2]);
    acc[1][2] = wmma_bf16(a1, b2, acc[1][2]);
    acc[0][3] = wmma_bf16(a0, b3, acc[0][3]);
    acc[1][3] = wmma_bf16(a1, b3, acc[1][3]);

    if (more) {
      wait_async0();
      __syncthreads();
    }
    cur ^= 1;
  }

  // C/D layout: lane l, vgpr r -> M = r + 8*(l>>4), N = l&15
  const int lane = tid & 31;
  const int nloc = lane & 15;
  const int halfw = lane >> 4;
  const float sc = (ep == EP_SCALAR || ep == EP_BASE_SCALAR) ? scalar[0] : 0.f;
  float* Cf = (float*)Cv;
  bf16*  Cb = (bf16*)Cv;
#pragma unroll
  for (int mt = 0; mt < 2; ++mt)
#pragma unroll
    for (int nt = 0; nt < 4; ++nt)
#pragma unroll
      for (int r = 0; r < 8; ++r) {
        int m = m0 + wm + mt * 16 + r + 8 * halfw;
        int n = n0 + wn + nt * 16 + nloc;
        if (n >= N) continue;
        size_t idx = (size_t)m * N + n;
        float v = acc[mt][nt][r];
        if      (ep == EP_SCALAR)      v *= sc;
        else if (ep == EP_BASE_SCALAR) v = base[idx] + sc * v;
        else if (ep == EP_BASE_COL)    v = base[idx] + colscale[n] * v;
        else if (ep == EP_ADD)         v = v + base[idx];
        else if (ep == EP_SQRELU)      { float t = v >= 0.f ? v : 0.5f * v; v = t * t; }
        if (obf) Cb[idx] = f2bf(v); else Cf[idx] = v;
      }
}

// ---- embed RMS gather:  E[row] = rms(embed_w[ids[row]]) ---------------------
__global__ __launch_bounds__(256)
void k_embed_rms(const int* __restrict__ ids, const float* __restrict__ embed_w,
                 float* __restrict__ E)
{
  __shared__ float red[8];
  int row = blockIdx.x;
  int tid = threadIdx.x;
  const float* src = embed_w + (size_t)ids[row] * Dc;
  float v[4]; float ss = 0.f;
#pragma unroll
  for (int i = 0; i < 4; ++i) { v[i] = src[tid + i * 256]; ss += v[i] * v[i]; }
#pragma unroll
  for (int off = 1; off < 32; off <<= 1) ss += __shfl_xor(ss, off, 32);
  if ((tid & 31) == 0) red[tid >> 5] = ss;
  __syncthreads();
  float tot = 0.f;
#pragma unroll
  for (int w = 0; w < 8; ++w) tot += red[w];
  float scl = rsqrtf(tot / (float)Dc + 1e-6f);
#pragma unroll
  for (int i = 0; i < 4; ++i) E[(size_t)row * Dc + tid + i * 256] = v[i] * scl;
}

// ---- hash + gather of big_w rows (bf16 out, GEMM A operand) -----------------
__global__ __launch_bounds__(256)
void k_gather_big(const int* __restrict__ ids, const float* __restrict__ big_w,
                  bf16* __restrict__ Gb)
{
  int idx = blockIdx.x * 256 + threadIdx.x;          // NTOK*BDc elements
  int row = idx >> 8;
  int c   = idx & 255;
  int b = row / Tc, t = row % Tc;
  int g;
  if (t == 0) g = 65535;
  else {
    int a = ids[b * Tc + t], p = ids[b * Tc + t - 1];
    int h = (int)(36313u * (unsigned)a ^ 27191u * (unsigned)p);
    int m = h % 65535; if (m < 0) m += 65535;
    g = m;
  }
  Gb[(size_t)row * BDc + c] = f2bf(big_w[(size_t)g * BDc + c]);
}

// ---- gather ve_w rows (bf16 out) -------------------------------------------
__global__ __launch_bounds__(256)
void k_gather_ve(const int* __restrict__ ids, const float* __restrict__ ve_w,
                 bf16* __restrict__ VEg)
{
  int idx = blockIdx.x * 256 + threadIdx.x;          // NTOK*VEDc
  int row = idx >> 7;
  int c   = idx & 127;
  VEg[(size_t)row * VEDc + c] = f2bf(ve_w[(size_t)ids[row] * VEDc + c]);
}

// ---- smear gate:  X = (1-g)*Xin + g*Xin_prev ; X0 = X -----------------------
__global__ __launch_bounds__(256)
void k_smear(const float* __restrict__ Xin, const float* __restrict__ gate,
             float* __restrict__ X, float* __restrict__ X0)
{
  size_t idx = (size_t)blockIdx.x * 256 + threadIdx.x;  // NTOK*Dc
  int d = (int)(idx & (Dc - 1));
  int t = (int)((idx / Dc) % Tc);
  float g = 1.f / (1.f + __expf(-gate[d]));
  float prev = (t > 0) ? Xin[idx - Dc] : 0.f;
  float v = (1.f - g) * Xin[idx] + g * prev;
  X[idx] = v; X0[idx] = v;
}

// ---- residual mix + RMS; XN emitted as bf16 (GEMM A operand) ----------------
__global__ __launch_bounds__(256)
void k_mix_rms(const float* __restrict__ X, const float* __restrict__ X0,
               const float* __restrict__ m0, const float* __restrict__ m1,
               float* __restrict__ XIN, bf16* __restrict__ XN)
{
  __shared__ float red[8];
  int row = blockIdx.x;
  int tid = threadIdx.x;
  const float* xr  = X + (size_t)row * Dc;
  const float* x0r = X0 ? X0 + (size_t)row * Dc : nullptr;
  float v[4]; float ss = 0.f;
#pragma unroll
  for (int i = 0; i < 4; ++i) {
    int d = tid + i * 256;
    float t = xr[d];
    if (m0) t = m0[d] * t + m1[d] * x0r[d];
    v[i] = t; ss += t * t;
  }
#pragma unroll
  for (int off = 1; off < 32; off <<= 1) ss += __shfl_xor(ss, off, 32);
  if ((tid & 31) == 0) red[tid >> 5] = ss;
  __syncthreads();
  float tot = 0.f;
#pragma unroll
  for (int w = 0; w < 8; ++w) tot += red[w];
  float scl = rsqrtf(tot / (float)Dc + 1e-6f);
#pragma unroll
  for (int i = 0; i < 4; ++i) {
    int d = tid + i * 256;
    if (XIN) XIN[(size_t)row * Dc + d] = v[i];
    XN[(size_t)row * Dc + d] = f2bf(v[i] * scl);
  }
}

// ---- per-head RMS + RoPE (+ q_gain for q heads) -----------------------------
__global__ __launch_bounds__(256)
void k_rope(float* __restrict__ Qb, float* __restrict__ Kb,
            const float* __restrict__ q_gain_l)
{
  int u = blockIdx.x * 256 + threadIdx.x;    // NTOK * 20
  int row = u / 20;
  int hh  = u % 20;
  int t = row % Tc;
  bool isq = hh < Hc;
  float* p = isq ? Qb + ((size_t)row * Hc + hh) * HDc
                 : Kb + ((size_t)row * HKVc + (hh - Hc)) * HDc;
  float vb[64]; float ss = 0.f;
#pragma unroll
  for (int d = 0; d < 64; ++d) { vb[d] = p[d]; ss += vb[d] * vb[d]; }
  float scl = rsqrtf(ss / 64.f + 1e-6f);
  float gain = isq ? q_gain_l[hh] : 1.f;
  const float LB = logf(10000.f) / 32.f;
#pragma unroll
  for (int i = 0; i < 32; ++i) {
    float ang = (float)t * __expf(-(float)i * LB);
    float c = __cosf(ang), s = __sinf(ang);
    float x1 = vb[i] * scl, x2 = vb[i + 32] * scl;
    p[i]      = (x1 * c + x2 * s) * gain;
    p[i + 32] = (-x1 * s + x2 * c) * gain;
  }
}

// ---- Flash attention (causal, GQA) with WMMA --------------------------------
// grid: (Tc/128, Hc, Bc); 256 threads = 8 waves; wave owns 16 query rows.
// K/V staging for tile kt+1 is pipelined against compute of tile kt.
#define TQ 128
#define TK 64
#define LQ 72   // padded bf16 stride

__global__ __launch_bounds__(256)
void k_attn(const float* __restrict__ Q, const float* __restrict__ Kb,
            const float* __restrict__ Vb, bf16* __restrict__ Yo)
{
  __shared__ __align__(16) char smem[(TQ + TK + HDc) * LQ * 2 + 8 * 16 * LQ * 2];
  bf16* Qs  = (bf16*)smem;                              // 128*72 bf16
  bf16* Ks  = (bf16*)(smem + TQ * LQ * 2);              // 64*72
  bf16* Vts = (bf16*)(smem + (TQ + TK) * LQ * 2);       // 64*72 (transposed V)
  bf16* Ps  = (bf16*)(smem + (TQ + TK + HDc) * LQ * 2); // 8 waves * 16*72

  const int q0   = blockIdx.x * TQ;
  const int hidx = blockIdx.y;
  const int bidx = blockIdx.z;
  const int kvh  = hidx / (Hc / HKVc);
  const int tid  = threadIdx.x;
  const int wave = tid >> 5;
  const int lane = tid & 31;
  const int nloc = lane & 15;
  const int halfw = lane >> 4;

  // stage Q tile (scaled by 1/sqrt(HD)); grouped loads then converts
  {
    float4 rq[8];
#pragma unroll
    for (int i = 0; i < 8; ++i) {
      int vid = tid + i * 256;        // 2048 float4s
      int row = vid >> 4;
      int c4  = (vid & 15) * 4;
      rq[i] = *(const float4*)(Q + (((size_t)bidx * Tc + q0 + row) * Hc + hidx) * HDc + c4);
    }
#pragma unroll
    for (int i = 0; i < 8; ++i) {
      int vid = tid + i * 256;
      int row = vid >> 4;
      int c4  = (vid & 15) * 4;
      bf16* d = Qs + row * LQ + c4;
      d[0] = f2bf(rq[i].x * 0.125f); d[1] = f2bf(rq[i].y * 0.125f);
      d[2] = f2bf(rq[i].z * 0.125f); d[3] = f2bf(rq[i].w * 0.125f);
    }
  }

  float4 rk[4], rv[4];
  auto load_kv = [&](int kt) {
#pragma unroll
    for (int i = 0; i < 4; ++i) {
      int vid = tid + i * 256;        // 1024 float4s (K tile 64x64)
      int row = vid >> 4;
      int c4  = (vid & 15) * 4;
      rk[i] = *(const float4*)(Kb + (((size_t)bidx * Tc + kt * TK + row) * HKVc + kvh) * HDc + c4);
    }
#pragma unroll
    for (int i = 0; i < 4; ++i) {
      int vid = tid + i * 256;        // V tile 64x64, coalesced along d
      int key = vid >> 4;
      int d4  = (vid & 15) * 4;
      rv[i] = *(const float4*)(Vb + (((size_t)bidx * Tc + kt * TK + key) * HKVc + kvh) * HDc + d4);
    }
  };
  auto store_kv = [&]() {
#pragma unroll
    for (int i = 0; i < 4; ++i) {
      int vid = tid + i * 256;
      int row = vid >> 4;
      int c4  = (vid & 15) * 4;
      bf16* d = Ks + row * LQ + c4;
      d[0] = f2bf(rk[i].x); d[1] = f2bf(rk[i].y);
      d[2] = f2bf(rk[i].z); d[3] = f2bf(rk[i].w);
    }
#pragma unroll
    for (int i = 0; i < 4; ++i) {     // transpose at the LDS-store side
      int vid = tid + i * 256;
      int key = vid >> 4;
      int d4  = (vid & 15) * 4;
      Vts[(d4 + 0) * LQ + key] = f2bf(rv[i].x);
      Vts[(d4 + 1) * LQ + key] = f2bf(rv[i].y);
      Vts[(d4 + 2) * LQ + key] = f2bf(rv[i].z);
      Vts[(d4 + 3) * LQ + key] = f2bf(rv[i].w);
    }
  };

  f32x8 yacc[4] = {};
  float m_run[8], l_run[8];
#pragma unroll
  for (int r = 0; r < 8; ++r) { m_run[r] = -1e30f; l_run[r] = 0.f; }

  const int nkt = (q0 + TQ) / TK;     // causal: keys up to q0+TQ-1
  load_kv(0);
  for (int kt = 0; kt < nkt; ++kt) {
    store_kv();
    __syncthreads();
    if (kt + 1 < nkt) load_kv(kt + 1);   // overlap next tile with compute

    // S = Q K^T   (16q x 64k per wave); all frags hoisted before wmma chain
    f32x8 sacc[4] = {};
    {
      bf16x16 a0 = load_fragA(Qs, wave * 16, LQ);
      bf16x16 a1 = load_fragA(Qs + 32, wave * 16, LQ);
      bf16x16 k00 = load_fragB(Ks, 0, LQ);
      bf16x16 k01 = load_fragB(Ks, 16, LQ);
      bf16x16 k02 = load_fragB(Ks, 32, LQ);
      bf16x16 k03 = load_fragB(Ks, 48, LQ);
      bf16x16 k10 = load_fragB(Ks + 32, 0, LQ);
      bf16x16 k11 = load_fragB(Ks + 32, 16, LQ);
      bf16x16 k12 = load_fragB(Ks + 32, 32, LQ);
      bf16x16 k13 = load_fragB(Ks + 32, 48, LQ);
      sacc[0] = wmma_bf16(a0, k00, sacc[0]);
      sacc[1] = wmma_bf16(a0, k01, sacc[1]);
      sacc[2] = wmma_bf16(a0, k02, sacc[2]);
      sacc[3] = wmma_bf16(a0, k03, sacc[3]);
      sacc[0] = wmma_bf16(a1, k10, sacc[0]);
      sacc[1] = wmma_bf16(a1, k11, sacc[1]);
      sacc[2] = wmma_bf16(a1, k12, sacc[2]);
      sacc[3] = wmma_bf16(a1, k13, sacc[3]);
    }

    // causal mask + online softmax (rows live across 16 lanes of a half-wave)
    float alpha[8];
#pragma unroll
    for (int r = 0; r < 8; ++r) {
      int mrow = q0 + wave * 16 + r + 8 * halfw;
      float mx = -1e30f;
#pragma unroll
      for (int nt = 0; nt < 4; ++nt) {
        int n = kt * TK + nt * 16 + nloc;
        float sv = sacc[nt][r];
        if (n > mrow) { sv = -1e30f; sacc[nt][r] = sv; }
        mx = fmaxf(mx, sv);
      }
#pragma unroll
      for (int off = 1; off < 16; off <<= 1) mx = fmaxf(mx, __shfl_xor(mx, off, 32));
      float mnew = fmaxf(m_run[r], mx);
      float al = __expf(m_run[r] - mnew);
      float lsum = 0.f;
#pragma unroll
      for (int nt = 0; nt < 4; ++nt) {
        float pv = __expf(sacc[nt][r] - mnew);
        sacc[nt][r] = pv; lsum += pv;
      }
#pragma unroll
      for (int off = 1; off < 16; off <<= 1) lsum += __shfl_xor(lsum, off, 32);
      l_run[r] = l_run[r] * al + lsum;
      m_run[r] = mnew;
      alpha[r] = al;
    }
#pragma unroll
    for (int nt = 0; nt < 4; ++nt)
#pragma unroll
      for (int r = 0; r < 8; ++r) yacc[nt][r] *= alpha[r];

    // P (C-layout) -> LDS -> A-fragments for P@V
    bf16* Pw = Ps + wave * 16 * LQ;
#pragma unroll
    for (int nt = 0; nt < 4; ++nt)
#pragma unroll
      for (int r = 0; r < 8; ++r)
        Pw[(r + 8 * halfw) * LQ + nt * 16 + nloc] = f2bf(sacc[nt][r]);

    {
      bf16x16 p0 = load_fragA(Pw, 0, LQ);
      bf16x16 p1 = load_fragA(Pw + 32, 0, LQ);
      bf16x16 v00 = load_fragB(Vts, 0, LQ);
      bf16x16 v01 = load_fragB(Vts, 16, LQ);
      bf16x16 v02 = load_fragB(Vts, 32, LQ);
      bf16x16 v03 = load_fragB(Vts, 48, LQ);
      bf16x16 v10 = load_fragB(Vts + 32, 0, LQ);
      bf16x16 v11 = load_fragB(Vts + 32, 16, LQ);
      bf16x16 v12 = load_fragB(Vts + 32, 32, LQ);
      bf16x16 v13 = load_fragB(Vts + 32, 48, LQ);
      yacc[0] = wmma_bf16(p0, v00, yacc[0]);
      yacc[1] = wmma_bf16(p0, v01, yacc[1]);
      yacc[2] = wmma_bf16(p0, v02, yacc[2]);
      yacc[3] = wmma_bf16(p0, v03, yacc[3]);
      yacc[0] = wmma_bf16(p1, v10, yacc[0]);
      yacc[1] = wmma_bf16(p1, v11, yacc[1]);
      yacc[2] = wmma_bf16(p1, v12, yacc[2]);
      yacc[3] = wmma_bf16(p1, v13, yacc[3]);
    }
    __syncthreads();
  }

  // normalize and park y in LDS (aliases Qs/Ks/Vts region; compute is done)
  float* yld = (float*)smem;          // 128 x 64 f32
#pragma unroll
  for (int nt = 0; nt < 4; ++nt)
#pragma unroll
    for (int r = 0; r < 8; ++r)
      yld[(wave * 16 + r + 8 * halfw) * HDc + nt * 16 + nloc] =
        yacc[nt][r] / l_run[r];
  __syncthreads();

  // per-row epilogue: remove component along normalized v(query position)
  if (tid < TQ) {
    int qrow = q0 + tid;
    const float* vp = Vb + (((size_t)bidx * Tc + qrow) * HKVc + kvh) * HDc;
    float nrm = 0.f;
#pragma unroll
    for (int d = 0; d < 64; ++d) { float t = vp[d]; nrm += t * t; }
    float inv = 1.f / fmaxf(sqrtf(nrm), 1e-12f);
    const float* yr = yld + tid * HDc;
    float dot = 0.f;
#pragma unroll
    for (int d = 0; d < 64; ++d) dot += yr[d] * (vp[d] * inv);
    bf16* op = Yo + (((size_t)bidx * Tc + qrow) * Hc + hidx) * HDc;
#pragma unroll
    for (int d = 0; d < 64; ++d) op[d] = f2bf(yr[d] - dot * (vp[d] * inv));
  }
}

// ---- host-side launchers ----------------------------------------------------
static inline void gemm(const bf16* A, const bf16* W, void* C,
                        int M, int N, int K, int ep, int obf,
                        const float* base, const float* colscale,
                        const float* scalar, hipStream_t s)
{
  dim3 g((N + GN - 1) / GN, M / GM);
  k_gemm<<<g, 256, 0, s>>>(A, W, C, M, N, K, ep, obf, base, colscale, scalar);
}
static inline void cvt(const float* w, bf16* dst, size_t n, hipStream_t s)
{
  long n4 = (long)(n / 4);
  k_cvt<<<(n4 + 255) / 256, 256, 0, s>>>(w, dst, n4);
}

extern "C" void kernel_launch(void* const* d_in, const int* in_sizes, int n_in,
                              void* d_out, int out_size, void* d_ws, size_t ws_size,
                              hipStream_t stream)
{
  const int*   ids        = (const int*)  d_in[0];
  const float* embed_w    = (const float*)d_in[1];
  const float* big_w      = (const float*)d_in[2];
  const float* big_proj   = (const float*)d_in[3];
  const float* big_scale  = (const float*)d_in[4];   // scalar
  const float* smear_gate = (const float*)d_in[5];
  const float* ve_w       = (const float*)d_in[6];
  const float* ve_proj    = (const float*)d_in[7];
  const float* ve_scale   = (const float*)d_in[8];   // scalar
  const float* Wq         = (const float*)d_in[9];
  const float* Wk         = (const float*)d_in[10];
  const float* Wv         = (const float*)d_in[11];
  const float* Wo         = (const float*)d_in[12];
  const float* q_gain     = (const float*)d_in[13];
  const float* attn_scale = (const float*)d_in[14];
  const float* mlp_scale  = (const float*)d_in[15];
  const float* resid_mix  = (const float*)d_in[16];
  const float* Wfc        = (const float*)d_in[17];
  const float* Wp         = (const float*)d_in[18];
  const float* head_w     = (const float*)d_in[19];
  float* out = (float*)d_out;

  // workspace carve (~175 MB): f32 activations + bf16 operands + weight scratch
  float* ws = (float*)d_ws;
  size_t off = 0;
  auto carveF = [&](size_t n) { float* p = ws + off; off += n; return p; };
  float* E    = carveF((size_t)NTOK * Dc);
  float* X    = carveF((size_t)NTOK * Dc);
  float* X0   = carveF((size_t)NTOK * Dc);
  float* XIN  = carveF((size_t)NTOK * Dc);
  float* Qb   = carveF((size_t)NTOK * Dc);
  float* Kbuf = carveF((size_t)NTOK * KVDc);
  float* Vbuf = carveF((size_t)NTOK * KVDc);
  float* VE   = carveF((size_t)NTOK * KVDc);
  bf16* bp = (bf16*)(ws + off);
  size_t boff = 0;
  auto carveB = [&](size_t n) { bf16* p = bp + boff; boff += n; return p; };
  bf16* XNb  = carveB((size_t)NTOK * Dc);
  bf16* Yb   = carveB((size_t)NTOK * Dc);
  bf16* Hb   = carveB((size_t)NTOK * MLPc);
  bf16* Gb   = carveB((size_t)NTOK * BDc);
  bf16* VEg  = carveB((size_t)NTOK * VEDc);
  bf16* WB   = carveB((size_t)Vc * Dc);      // bf16 weight scratch (max: head)
  (void)ws_size; (void)in_sizes; (void)n_in; (void)out_size;

  // --- embedding path ---
  k_embed_rms<<<NTOK, 256, 0, stream>>>(ids, embed_w, E);
  k_gather_big<<<(NTOK * BDc) / 256, 256, 0, stream>>>(ids, big_w, Gb);
  k_gather_ve<<<(NTOK * VEDc) / 256, 256, 0, stream>>>(ids, ve_w, VEg);
  cvt(big_proj, WB, (size_t)Dc * BDc, stream);
  gemm(Gb, WB, XIN, NTOK, Dc, BDc, EP_BASE_SCALAR, 0, E, nullptr, big_scale, stream);
  k_smear<<<(NTOK * Dc) / 256, 256, 0, stream>>>(XIN, smear_gate, X, X0);
  cvt(ve_proj, WB, (size_t)KVDc * VEDc, stream);
  gemm(VEg, WB, VE, NTOK, KVDc, VEDc, EP_SCALAR, 0, nullptr, nullptr, ve_scale, stream);

  // --- transformer layers ---
  for (int l = 0; l < Lc; ++l) {
    const float* m0 = resid_mix + ((size_t)l * 2 + 0) * Dc;
    const float* m1 = resid_mix + ((size_t)l * 2 + 1) * Dc;
    k_mix_rms<<<NTOK, 256, 0, stream>>>(X, X0, m0, m1, XIN, XNb);

    cvt(Wq + (size_t)l * Dc * Dc, WB, (size_t)Dc * Dc, stream);
    gemm(XNb, WB, Qb, NTOK, Dc, Dc, EP_NONE, 0, nullptr, nullptr, nullptr, stream);
    cvt(Wk + (size_t)l * KVDc * Dc, WB, (size_t)KVDc * Dc, stream);
    gemm(XNb, WB, Kbuf, NTOK, KVDc, Dc, EP_NONE, 0, nullptr, nullptr, nullptr, stream);
    cvt(Wv + (size_t)l * KVDc * Dc, WB, (size_t)KVDc * Dc, stream);
    gemm(XNb, WB, Vbuf, NTOK, KVDc, Dc, EP_ADD, 0, VE, nullptr, nullptr, stream);

    k_rope<<<(NTOK * (Hc + HKVc)) / 256, 256, 0, stream>>>(Qb, Kbuf, q_gain + (size_t)l * Hc);

    k_attn<<<dim3(Tc / TQ, Hc, Bc), 256, 0, stream>>>(Qb, Kbuf, Vbuf, Yb);

    cvt(Wo + (size_t)l * Dc * Dc, WB, (size_t)Dc * Dc, stream);
    gemm(Yb, WB, X, NTOK, Dc, Dc, EP_BASE_COL, 0,
         XIN, attn_scale + (size_t)l * Dc, nullptr, stream);

    k_mix_rms<<<NTOK, 256, 0, stream>>>(X, nullptr, nullptr, nullptr, nullptr, XNb);
    cvt(Wfc + (size_t)l * MLPc * Dc, WB, (size_t)MLPc * Dc, stream);
    gemm(XNb, WB, Hb, NTOK, MLPc, Dc, EP_SQRELU, 1, nullptr, nullptr, nullptr, stream);
    cvt(Wp + (size_t)l * Dc * MLPc, WB, (size_t)Dc * MLPc, stream);
    gemm(Hb, WB, X, NTOK, Dc, MLPc, EP_BASE_COL, 0,
         X, mlp_scale + (size_t)l * Dc, nullptr, stream);
  }

  // --- head ---
  k_mix_rms<<<NTOK, 256, 0, stream>>>(X, nullptr, nullptr, nullptr, nullptr, XNb);
  cvt(head_w, WB, (size_t)Vc * Dc, stream);
  gemm(XNb, WB, out, NTOK, Vc, Dc, EP_NONE, 0, nullptr, nullptr, nullptr, stream);
}
